// Regressor_77017353552198
// MI455X (gfx1250) — matrix operand order repaired and checked by reading, not compiled, not run on hardware
//
#include <hip/hip_runtime.h>
#include <hip/hip_bf16.h>
#include <math.h>

#define NP 4096
#define IMG 96
#define NPIX (IMG * IMG)
#define ZD 64

typedef __attribute__((ext_vector_type(16))) _Float16 v16h;
typedef __attribute__((ext_vector_type(8)))  float    v8f;
typedef unsigned int u32;
typedef u32 u32x4 __attribute__((ext_vector_type(4)));
typedef u32 u32x8 __attribute__((ext_vector_type(8)));

// ---------------------------------------------------------------------------
// WMMA GEMM:  Y[O,N] = W[O,C] * X[C,N] + b[O]   (optional leaky-relu)
// Per block: 128 output cols, 16 output rows. Per K-chunk of 32:
//   wave 0 issues one TDM tensor_load_to_lds for the 32 x tw f32 X-tile
//   (2D descriptor: tensor {N x C}, stride N, tile {tw x th}), waits
//   s_wait_tensorcnt, then a barrier publishes the tile. Waves read B
//   fragments from LDS, A fragments from global (contiguous per lane),
//   and issue two v_wmma_f32_16x16x32_f16 (two 16-col tiles share A).
// NOTE: all callers use N % 16 == 0.
// ---------------------------------------------------------------------------
__global__ __launch_bounds__(128) void k_gemm(const float* __restrict__ W,
                                              const float* __restrict__ X,
                                              const float* __restrict__ B,
                                              float* __restrict__ Y,
                                              int O, int C, int N, int act)
{
    __shared__ float tileB[32 * 128];     // TDM destination: [K-rows][cols]
    int wave = threadIdx.x >> 5;
    int lane = threadIdx.x & 31;
    int nblock = blockIdx.x * 128;
    int n0 = nblock + wave * 32;
    int o0 = blockIdx.y * 16;
    int row  = lane & 15;
    int half = lane >> 4;
    bool active = (n0 < N);               // wave-uniform
    bool fullO = (o0 + 16 <= O);
    bool tile1 = active && (n0 + 32 <= N);
    int tw = N - nblock; if (tw > 128) tw = 128;   // tile width (cols)

    const float* Wr = W + (size_t)(o0 + row) * C;
    int colB0 = wave * 32 + row;          // tile0 col within LDS tile

    v8f acc0 = {}, acc1 = {};
    for (int c0 = 0; c0 < C; c0 += 32) {
        int th = C - c0; if (th > 32) th = 32;     // tile height (K rows)
        bool fullK = (th == 32);

        __syncthreads();                  // previous chunk's LDS reads done
        if (wave == 0) {
            // ---- Tensor DMA: X[c0:c0+th, nblock:nblock+tw] -> LDS
            u32 lds = (u32)(size_t)&tileB[0];      // flat addr low 32 = LDS offset
            unsigned long long ga =
                (unsigned long long)(size_t)(X + (size_t)c0 * N + nblock);
            u32x4 g0;
            g0.x = 1u;                                            // count=1
            g0.y = lds;                                           // lds_addr
            g0.z = (u32)ga;                                       // global_addr lo
            g0.w = (u32)((ga >> 32) & 0x01FFFFFFu) | (2u << 30);  // addr hi | type=2
            u32x8 g1;
            g1.s0 = 2u << 16;                                     // data_size = 4B
            g1.s1 = ((u32)N & 0xFFFFu) << 16;                     // tensor_dim0 lo16
            g1.s2 = (((u32)N >> 16) & 0xFFFFu) |
                    (((u32)C & 0xFFFFu) << 16);                   // dim0 hi | dim1 lo
            g1.s3 = (((u32)C >> 16) & 0xFFFFu) |
                    ((u32)tw << 16);                              // dim1 hi | tile_dim0
            g1.s4 = (u32)th;                                      // tile_dim1 (dim2=0)
            g1.s5 = (u32)N;                                       // dim0_stride lo32
            g1.s6 = 0u;                                           // stride hi | dim1_stride lo
            g1.s7 = 0u;                                           // dim1_stride hi
            asm volatile("tensor_load_to_lds %0, %1"
                         :: "s"(g0), "s"(g1) : "memory");
            __builtin_amdgcn_s_wait_tensorcnt(0);
        }
        __syncthreads();                  // publish tile to all waves

        if (active) {
            v16h a, b0, b1;
            // ---- A fragment (16x32 f16 layout): two runs of 8 consecutive K
            if (fullK && fullO) {
                const float* p0 = Wr + c0 + half * 8;
                const float* p1 = p0 + 16;
#pragma unroll
                for (int e = 0; e < 8; ++e) a[e] = (_Float16)p0[e];
#pragma unroll
                for (int e = 0; e < 8; ++e) a[8 + e] = (_Float16)p1[e];
            } else {
#pragma unroll
                for (int v = 0; v < 8; ++v)
#pragma unroll
                    for (int e = 0; e < 2; ++e) {
                        int ka = ((v < 4) ? 0 : 16) + half * 8 + ((v & 3) << 1) + e;
                        int c = c0 + ka;
                        float wa = 0.f;
                        if (c < C && (o0 + row) < O) wa = Wr[c];
                        a[v * 2 + e] = (_Float16)wa;
                    }
            }
            // ---- B fragments from the TDM-staged LDS tile: b[i] <-> K = (i<8?0:16)+half*8+(i&7)
            if (fullK) {
#pragma unroll
                for (int i = 0; i < 16; ++i)
                    b0[i] = (_Float16)tileB[(((i < 8) ? 0 : 16) + half * 8 + (i & 7)) * tw + colB0];
                if (tile1) {
#pragma unroll
                    for (int i = 0; i < 16; ++i)
                        b1[i] = (_Float16)tileB[(((i < 8) ? 0 : 16) + half * 8 + (i & 7)) * tw + colB0 + 16];
                }
            } else {
#pragma unroll
                for (int i = 0; i < 16; ++i) {
                    int k = ((i < 8) ? 0 : 16) + half * 8 + (i & 7);
                    float x0 = 0.f, x1 = 0.f;
                    if (k < th) {
                        x0 = tileB[k * tw + colB0];
                        if (tile1) x1 = tileB[k * tw + colB0 + 16];
                    }
                    b0[i] = (_Float16)x0;
                    b1[i] = (_Float16)x1;
                }
            }
            acc0 = __builtin_amdgcn_wmma_f32_16x16x32_f16(false, a, false, b0,
                                                          (short)0, acc0, false, false);
            if (tile1)
                acc1 = __builtin_amdgcn_wmma_f32_16x16x32_f16(false, a, false, b1,
                                                              (short)0, acc1, false, false);
        }
    }

    if (active) {
#pragma unroll
        for (int v = 0; v < 8; ++v) {
            int m = half * 8 + v;         // C/D layout: VGPR v -> row v (+8 hi lanes)
            int oo = o0 + m;
            if (oo < O) {
                float bb = B[oo];
                float r = acc0[v] + bb;
                if (act == 1) r = (r > 0.f) ? r : 0.1f * r;
                Y[(size_t)oo * N + n0 + row] = r;
                if (tile1) {
                    float r1 = acc1[v] + bb;
                    if (act == 1) r1 = (r1 > 0.f) ? r1 : 0.1f * r1;
                    Y[(size_t)oo * N + n0 + 16 + row] = r1;
                }
            }
        }
    }
}

// ---------------------------------------------------------------------------
// GroupNorm (8 groups) + leaky-relu, in place. One block per group.
// ---------------------------------------------------------------------------
__global__ __launch_bounds__(256) void k_gn_lrelu(float* __restrict__ Y, int C, int N)
{
    const int groups = 8;
    int g = blockIdx.x;
    int cpg = C / groups;
    long total = (long)cpg * N;
    __shared__ float ss[256], sq[256];
    int tid = threadIdx.x;
    float s = 0.f, q = 0.f;
    for (long i = tid; i < total; i += 256) {
        int c = g * cpg + (int)(i / N);
        int n = (int)(i % N);
        float v = Y[(size_t)c * N + n];
        s += v; q += v * v;
    }
    ss[tid] = s; sq[tid] = q;
    __syncthreads();
    for (int off = 128; off > 0; off >>= 1) {
        if (tid < off) { ss[tid] += ss[tid + off]; sq[tid] += sq[tid + off]; }
        __syncthreads();
    }
    float mu  = ss[0] / (float)total;
    float var = sq[0] / (float)total - mu * mu;
    float inv = rsqrtf(var + 1e-5f);
    for (long i = tid; i < total; i += 256) {
        int c = g * cpg + (int)(i / N);
        int n = (int)(i % N);
        float v = (Y[(size_t)c * N + n] - mu) * inv;
        Y[(size_t)c * N + n] = (v > 0.f) ? v : 0.1f * v;
    }
}

// ---------------------------------------------------------------------------
// SA helpers: center selection, kNN (top-K ascending + radius fallback),
// grouping (feat gather + relative xyz), max-pool over K.
// ---------------------------------------------------------------------------
__global__ void k_centers(const float* __restrict__ xyz, float* __restrict__ cxyz,
                          int N, int M)
{
    int m = blockIdx.x * blockDim.x + threadIdx.x;
    if (m >= M) return;
    int ci = (int)(((float)m * (float)(N - 1)) / (float)(M - 1));
    if (ci > N - 1) ci = N - 1;
    cxyz[0 * M + m] = xyz[0 * N + ci];
    cxyz[1 * M + m] = xyz[1 * N + ci];
    cxyz[2 * M + m] = xyz[2 * N + ci];
}

__global__ __launch_bounds__(256) void k_knn(const float* __restrict__ xyz,
                                             const float* __restrict__ cxyz,
                                             int* __restrict__ idx,
                                             int N, int M, int K, float r2)
{
    __shared__ float d2[4096];
    __shared__ float rmin[256];
    __shared__ int   rarg[256];
    __shared__ int   sIdx[32];
    __shared__ float sD[32];
    int m = blockIdx.x;
    int tid = threadIdx.x;
    float cx = cxyz[0 * M + m], cy = cxyz[1 * M + m], cz = cxyz[2 * M + m];
    for (int i = tid; i < N; i += 256) {
        float dx = xyz[0 * N + i] - cx;
        float dy = xyz[1 * N + i] - cy;
        float dz = xyz[2 * N + i] - cz;
        d2[i] = dx * dx + dy * dy + dz * dz;
    }
    __syncthreads();
    for (int k = 0; k < K; ++k) {
        float best = 3.4e38f; int bi = 0;
        for (int i = tid; i < N; i += 256)
            if (d2[i] < best) { best = d2[i]; bi = i; }
        rmin[tid] = best; rarg[tid] = bi;
        __syncthreads();
        for (int off = 128; off > 0; off >>= 1) {
            if (tid < off) {
                if (rmin[tid + off] < rmin[tid] ||
                    (rmin[tid + off] == rmin[tid] && rarg[tid + off] < rarg[tid])) {
                    rmin[tid] = rmin[tid + off]; rarg[tid] = rarg[tid + off];
                }
            }
            __syncthreads();
        }
        if (tid == 0) { sIdx[k] = rarg[0]; sD[k] = rmin[0]; d2[rarg[0]] = 3.4e38f; }
        __syncthreads();
    }
    if (tid < K) {
        int v = sIdx[tid];
        if (sD[tid] > r2) v = sIdx[0];    // radius fallback to nearest
        idx[m * K + tid] = v;
    }
}

__global__ void k_group(const float* __restrict__ feat, const float* __restrict__ xyz,
                        const float* __restrict__ cxyz, const int* __restrict__ idx,
                        float* __restrict__ out, int C, int N, int M, int K)
{
    int t = blockIdx.x * blockDim.x + threadIdx.x;
    int MK = M * K;
    int total = (C + 3) * MK;
    if (t >= total) return;
    int c = t / MK, col = t % MK;
    int m = col / K, k = col % K;
    int p = idx[m * K + k];
    float v;
    if (c < C) v = feat[(size_t)c * N + p];
    else { int d = c - C; v = xyz[d * N + p] - cxyz[d * M + m]; }
    out[(size_t)c * MK + col] = v;
}

__global__ void k_maxpool(const float* __restrict__ in, float* __restrict__ out,
                          int C, int M, int K)
{
    int t = blockIdx.x * blockDim.x + threadIdx.x;
    if (t >= C * M) return;
    int c = t / M, m = t % M;
    float mx = -3.4e38f;
    for (int k = 0; k < K; ++k)
        mx = fmaxf(mx, in[(size_t)c * M * K + m * K + k]);
    out[(size_t)c * M + m] = mx;
}

// ---------------------------------------------------------------------------
// FP helpers: 3-NN interpolation
// ---------------------------------------------------------------------------
__global__ void k_knn3(const float* __restrict__ dxyz, const float* __restrict__ sxyz,
                       int* __restrict__ idx, float* __restrict__ w, int Nd, int Ns)
{
    int n = blockIdx.x * blockDim.x + threadIdx.x;
    if (n >= Nd) return;
    float x = dxyz[0 * Nd + n], y = dxyz[1 * Nd + n], z = dxyz[2 * Nd + n];
    float b0 = 3.4e38f, b1 = 3.4e38f, b2 = 3.4e38f;
    int i0 = 0, i1 = 0, i2 = 0;
    for (int s = 0; s < Ns; ++s) {
        float dx = sxyz[0 * Ns + s] - x;
        float dy = sxyz[1 * Ns + s] - y;
        float dz = sxyz[2 * Ns + s] - z;
        float d = dx * dx + dy * dy + dz * dz;
        if (d < b0)      { b2 = b1; i2 = i1; b1 = b0; i1 = i0; b0 = d; i0 = s; }
        else if (d < b1) { b2 = b1; i2 = i1; b1 = d; i1 = s; }
        else if (d < b2) { b2 = d; i2 = s; }
    }
    float w0 = 1.f / (b0 + 1e-8f), w1 = 1.f / (b1 + 1e-8f), w2 = 1.f / (b2 + 1e-8f);
    float s = w0 + w1 + w2;
    idx[n * 3 + 0] = i0; idx[n * 3 + 1] = i1; idx[n * 3 + 2] = i2;
    w[n * 3 + 0] = w0 / s; w[n * 3 + 1] = w1 / s; w[n * 3 + 2] = w2 / s;
}

__global__ void k_interp(const float* __restrict__ sfeat, const int* __restrict__ idx,
                         const float* __restrict__ w, float* __restrict__ out,
                         int C, int Nd, int Ns)
{
    int t = blockIdx.x * blockDim.x + threadIdx.x;
    if (t >= C * Nd) return;
    int c = t / Nd, n = t % Nd;
    float s = 0.f;
    for (int k = 0; k < 3; ++k)
        s += sfeat[(size_t)c * Ns + idx[n * 3 + k]] * w[n * 3 + k];
    out[(size_t)c * Nd + n] = s;
}

__global__ void k_rowcopy(const float* __restrict__ src, float* __restrict__ dst, int n)
{
    int t = blockIdx.x * blockDim.x + threadIdx.x;
    if (t < n) dst[t] = src[t];
}

__global__ void k_add(const float* __restrict__ a, const float* __restrict__ b,
                      float* __restrict__ o, int n)
{
    int t = blockIdx.x * blockDim.x + threadIdx.x;
    if (t < n) o[t] = a[t] + b[t];
}

// ---------------------------------------------------------------------------
// Attention (H=8, D=16, N=16). qkv: 384x16 already computed by WMMA GEMM.
// ---------------------------------------------------------------------------
__global__ __launch_bounds__(256) void k_attention(const float* __restrict__ qkv,
                                                   float* __restrict__ out)
{
    __shared__ float sq[128][16], sk[128][16], sv[128][16];
    __shared__ float sctx[8][16][16];
    int tid = threadIdx.x;
    for (int t = tid; t < 128 * 16; t += 256) {
        int r = t >> 4, n = t & 15;
        sq[r][n] = qkv[r * 16 + n] * 0.25f;          // D^-0.5
        sk[r][n] = qkv[(128 + r) * 16 + n];
        sv[r][n] = qkv[(256 + r) * 16 + n];
    }
    __syncthreads();
    if (tid < 128) {                                  // softmax over n per (h,d)
        float mx = -3.4e38f;
        for (int n = 0; n < 16; ++n) mx = fmaxf(mx, sk[tid][n]);
        float s = 0.f;
        for (int n = 0; n < 16; ++n) { float e = __expf(sk[tid][n] - mx); sk[tid][n] = e; s += e; }
        for (int n = 0; n < 16; ++n) sk[tid][n] /= s;
    }
    __syncthreads();
    for (int t = tid; t < 8 * 16 * 16; t += 256) {    // ctx[h,d,e]
        int h = t >> 8, d = (t >> 4) & 15, e = t & 15;
        float s = 0.f;
        for (int n = 0; n < 16; ++n) s += sk[h * 16 + d][n] * sv[h * 16 + e][n];
        sctx[h][d][e] = s;
    }
    __syncthreads();
    for (int t = tid; t < 128 * 16; t += 256) {       // out[h,e,n]
        int r = t >> 4, n = t & 15;
        int h = r >> 4, e = r & 15;
        float s = 0.f;
        for (int d = 0; d < 16; ++d) s += sctx[h][d][e] * sq[h * 16 + d][n];
        out[r * 16 + n] = s;
    }
}

// ---------------------------------------------------------------------------
// Head post-processing: scale softplus clamp, quat normalize, opacity sigmoid
// ---------------------------------------------------------------------------
__global__ void k_post(const float* __restrict__ sraw, const float* __restrict__ rraw,
                       const float* __restrict__ oraw, float* __restrict__ scale,
                       float* __restrict__ rot, float* __restrict__ opac, int N)
{
    int n = blockIdx.x * blockDim.x + threadIdx.x;
    if (n >= N) return;
    for (int d = 0; d < 3; ++d) {
        float t = 100.f * sraw[d * N + n];
        float sp = (t > 20.f) ? t : log1pf(expf(t));
        scale[d * N + n] = fminf(sp * 0.01f, 0.01f);
    }
    float q0 = rraw[0 * N + n], q1 = rraw[1 * N + n], q2 = rraw[2 * N + n], q3 = rraw[3 * N + n];
    float nrm = sqrtf(q0 * q0 + q1 * q1 + q2 * q2 + q3 * q3) + 1e-8f;
    rot[0 * N + n] = q0 / nrm; rot[1 * N + n] = q1 / nrm;
    rot[2 * N + n] = q2 / nrm; rot[3 * N + n] = q3 / nrm;
    opac[n] = 1.f / (1.f + __expf(-oraw[n]));
}

// ---------------------------------------------------------------------------
// Gaussian projection: per-gaussian 2D cov inverse + screen pos + depth
// gp layout per gaussian: {u, v, ia, ib, ic, alpha_coef, z, pad}
// ---------------------------------------------------------------------------
__global__ void k_project(const float* __restrict__ pc, const float* __restrict__ scale,
                          const float* __restrict__ rot, const float* __restrict__ opac,
                          const float* __restrict__ vm, const float* __restrict__ intr,
                          float* __restrict__ gp, float* __restrict__ zk, int N)
{
    int n = blockIdx.x * blockDim.x + threadIdx.x;
    if (n >= N) return;
    float fx = intr[0], fy = intr[1], cx = intr[2], cy = intr[3];
    float Rc[3][3] = {{vm[0], vm[1], vm[2]}, {vm[4], vm[5], vm[6]}, {vm[8], vm[9], vm[10]}};
    float tx = vm[3], ty = vm[7], tz = vm[11];
    float X = pc[0 * N + n], Yp = pc[1 * N + n], Z = pc[2 * N + n];
    float px = Rc[0][0] * X + Rc[0][1] * Yp + Rc[0][2] * Z + tx;
    float py = Rc[1][0] * X + Rc[1][1] * Yp + Rc[1][2] * Z + ty;
    float pz = Rc[2][0] * X + Rc[2][1] * Yp + Rc[2][2] * Z + tz;
    float z = fmaxf(pz, 1e-4f);
    float u = fx * px / z + cx, v = fy * py / z + cy;
    float qw = rot[0 * N + n], qx = rot[1 * N + n], qy = rot[2 * N + n], qz = rot[3 * N + n];
    float Rq[3][3] = {
        {1.f - 2.f * (qy * qy + qz * qz), 2.f * (qx * qy - qw * qz), 2.f * (qx * qz + qw * qy)},
        {2.f * (qx * qy + qw * qz), 1.f - 2.f * (qx * qx + qz * qz), 2.f * (qy * qz - qw * qx)},
        {2.f * (qx * qz - qw * qy), 2.f * (qy * qz + qw * qx), 1.f - 2.f * (qx * qx + qy * qy)}};
    float sc[3] = {scale[0 * N + n], scale[1 * N + n], scale[2 * N + n]};
    float Mm[3][3], cov3[3][3], tmp[3][3], covc[3][3];
    for (int i = 0; i < 3; ++i)
        for (int j = 0; j < 3; ++j) Mm[i][j] = Rq[i][j] * sc[j];
    for (int i = 0; i < 3; ++i)
        for (int j = 0; j < 3; ++j) {
            float s = 0.f;
            for (int k = 0; k < 3; ++k) s += Mm[i][k] * Mm[j][k];
            cov3[i][j] = s;
        }
    for (int i = 0; i < 3; ++i)
        for (int j = 0; j < 3; ++j) {
            float s = 0.f;
            for (int k = 0; k < 3; ++k) s += Rc[i][k] * cov3[k][j];
            tmp[i][j] = s;
        }
    for (int i = 0; i < 3; ++i)
        for (int j = 0; j < 3; ++j) {
            float s = 0.f;
            for (int k = 0; k < 3; ++k) s += tmp[i][k] * Rc[j][k];
            covc[i][j] = s;
        }
    float J[2][3] = {{fx / z, 0.f, -fx * px / (z * z)}, {0.f, fy / z, -fy * py / (z * z)}};
    float t2[2][3], cov2[2][2];
    for (int i = 0; i < 2; ++i)
        for (int j = 0; j < 3; ++j) {
            float s = 0.f;
            for (int k = 0; k < 3; ++k) s += J[i][k] * covc[k][j];
            t2[i][j] = s;
        }
    for (int i = 0; i < 2; ++i)
        for (int j = 0; j < 2; ++j) {
            float s = 0.f;
            for (int k = 0; k < 3; ++k) s += t2[i][k] * J[j][k];
            cov2[i][j] = s;
        }
    float a = cov2[0][0] + 0.3f, b = cov2[0][1], c = cov2[1][1] + 0.3f;
    float det = a * c - b * b;
    int valid = (pz > 0.1f) && (det > 1e-9f);
    gp[n * 8 + 0] = u; gp[n * 8 + 1] = v;
    gp[n * 8 + 2] = c / det; gp[n * 8 + 3] = -b / det; gp[n * 8 + 4] = a / det;
    gp[n * 8 + 5] = valid ? opac[n] : 0.f;
    gp[n * 8 + 6] = z; gp[n * 8 + 7] = 0.f;
    zk[n] = z;
}

// ---------------------------------------------------------------------------
// Bitonic sort by z ascending (N=4096), one block, LDS resident.
// ---------------------------------------------------------------------------
__global__ __launch_bounds__(1024) void k_sortz(const float* __restrict__ zk,
                                                int* __restrict__ order, int N)
{
    __shared__ float key[4096];
    __shared__ int   val[4096];
    int tid = threadIdx.x;
    for (int i = tid; i < N; i += 1024) { key[i] = zk[i]; val[i] = i; }
    __syncthreads();
    for (int size = 2; size <= N; size <<= 1) {
        for (int stride = size >> 1; stride > 0; stride >>= 1) {
            for (int i = tid; i < N; i += 1024) {
                int j = i ^ stride;
                if (j > i) {
                    bool up = ((i & size) == 0);
                    if ((key[i] > key[j]) == up) {
                        float tk = key[i]; key[i] = key[j]; key[j] = tk;
                        int   tv = val[i]; val[i] = val[j]; val[j] = tv;
                    }
                }
            }
            __syncthreads();
        }
    }
    for (int i = tid; i < N; i += 1024) order[i] = val[i];
}

// ---------------------------------------------------------------------------
// Alpha compositing: 1 thread = 1 pixel; gaussians staged in LDS chunks of 64.
// Per-thread state: T, rgb[3], 64 feature accumulators (registers).
// ---------------------------------------------------------------------------
#define GCHUNK 64
__global__ __launch_bounds__(256) void k_composite(const float* __restrict__ gp,
                                                   const int* __restrict__ order,
                                                   const float* __restrict__ pc,
                                                   const float* __restrict__ feat,
                                                   float* __restrict__ rgb_out,
                                                   float* __restrict__ fmap_out,
                                                   int N)
{
    __shared__ float sg[GCHUNK][8];
    __shared__ float sc[GCHUNK][4];
    __shared__ float sf[GCHUNK][ZD];
    int px = blockIdx.x * blockDim.x + threadIdx.x;
    float x = (float)(px % IMG) + 0.5f;
    float y = (float)(px / IMG) + 0.5f;
    float T = 1.f, aR = 0.f, aG = 0.f, aB = 0.f;
    float aF[ZD];
#pragma unroll
    for (int c = 0; c < ZD; ++c) aF[c] = 0.f;

    for (int base = 0; base < N; base += GCHUNK) {
        __syncthreads();
        for (int t = threadIdx.x; t < GCHUNK * 8; t += 256) {
            int g = t >> 3, f = t & 7;
            sg[g][f] = gp[(size_t)order[base + g] * 8 + f];
        }
        for (int t = threadIdx.x; t < GCHUNK * 3; t += 256) {
            int g = t / 3, d = t % 3;
            sc[g][d] = pc[(size_t)(3 + d) * N + order[base + g]];
        }
        for (int t = threadIdx.x; t < GCHUNK * ZD; t += 256) {
            int g = t / ZD, c = t % ZD;
            sf[g][c] = feat[(size_t)c * N + order[base + g]];
        }
        __syncthreads();
        for (int g = 0; g < GCHUNK; ++g) {
            float dx = x - sg[g][0], dy = y - sg[g][1];
            float pw = -0.5f * (sg[g][2] * dx * dx + 2.f * sg[g][3] * dx * dy + sg[g][4] * dy * dy);
            pw = fminf(pw, 0.f);
            float alpha = fminf(0.99f, sg[g][5] * __expf(pw));
            float w = alpha * T;
            aR += w * sc[g][0]; aG += w * sc[g][1]; aB += w * sc[g][2];
#pragma unroll
            for (int c = 0; c < ZD; ++c) aF[c] += w * sf[g][c];
            T *= (1.f - alpha);
        }
    }
    rgb_out[0 * NPIX + px] = aR + T;      // bg = 1
    rgb_out[1 * NPIX + px] = aG + T;
    rgb_out[2 * NPIX + px] = aB + T;
#pragma unroll
    for (int c = 0; c < ZD; ++c) fmap_out[(size_t)c * NPIX + px] = aF[c];
}

// ---------------------------------------------------------------------------
// Host-side orchestration
// ---------------------------------------------------------------------------
extern "C" void kernel_launch(void* const* d_in, const int* in_sizes, int n_in,
                              void* d_out, int out_size, void* d_ws, size_t ws_size,
                              hipStream_t stream)
{
    (void)in_sizes; (void)n_in; (void)out_size; (void)ws_size;
    const float* pc   = (const float*)d_in[0];   // [1,6,4096]
    const float* vm   = (const float*)d_in[1];   // 4x4
    const float* intr = (const float*)d_in[2];   // 4
    const float* rgbp = pc + 3 * NP;

    // --- params, in jax pytree flatten order (dict keys sorted) ---
    int pi = 3;
    struct Lin { const float* w; const float* b; };
    auto nx  = [&]() { return (const float*)d_in[pi++]; };
    auto lin = [&]() { Lin l; l.w = nx(); l.b = nx(); return l; };

    Lin att_out = lin(), att_qkv = lin();               // att: out, qkv
    Lin cls0 = lin(), cls1 = lin();                     // cls
    const int fp_nconv[4] = {3, 3, 2, 2};
    const int fp_nmlp[4]  = {2, 2, 2, 3};
    Lin fpc[4][3], fpm[4][3];
    for (int j = 0; j < 4; ++j) {                       // fp: convs then mlp
        for (int k = 0; k < fp_nconv[j]; ++k) fpc[j][k] = lin();
        for (int k = 0; k < fp_nmlp[j]; ++k)  fpm[j][k] = lin();
    }
    Lin op0 = lin(), op1 = lin();                       // opac_head
    Lin ro0 = lin(), ro1 = lin();                       // rot_head
    const int sa_nconv[4] = {2, 3, 3, 0};
    const int sa_nmlp[4]  = {2, 2, 2, 3};
    Lin sac[4][3], sam[4][3];
    for (int i = 0; i < 4; ++i) {                       // sa: convs then mlp
        for (int k = 0; k < sa_nconv[i]; ++k) sac[i][k] = lin();
        for (int k = 0; k < sa_nmlp[i]; ++k)  sam[i][k] = lin();
    }
    Lin sc0 = lin(), sc1 = lin();                       // scale_head

    // --- workspace bump allocator ---
    char* wp = (char*)d_ws;
    auto alloc = [&](size_t nfl) {
        float* p = (float*)wp;
        wp += ((nfl * sizeof(float) + 255) & ~(size_t)255);
        return p;
    };
    float* T0 = alloc(2200000);
    float* T1 = alloc(2200000);
    float* T2 = alloc(2200000);
    float* pooled[4] = {alloc(64 * 1024), alloc(128 * 256), alloc(256 * 64), alloc(128 * 16)};
    float* c1 = alloc(3 * 1024); float* c2 = alloc(3 * 256);
    float* c3 = alloc(3 * 64);   float* c4 = alloc(3 * 16);
    int* idxg[4] = {(int*)alloc(1024 * 32), (int*)alloc(256 * 32),
                    (int*)alloc(64 * 32),   (int*)alloc(16 * 32)};
    int*   idx3 = (int*)alloc(NP * 3);
    float* w3   = alloc(NP * 3);
    float* attF = alloc(128 * 16);
    float* featZ = alloc((size_t)ZD * NP);
    float* sraw = alloc(3 * NP); float* rraw = alloc(4 * NP); float* oraw = alloc(NP);
    float* scaleB = alloc(3 * NP); float* rotB = alloc(4 * NP); float* opacB = alloc(NP);
    float* gp = alloc(NP * 8); float* zk = alloc(NP);
    int* order = (int*)alloc(NP);

    auto cdiv = [](int a, int b) { return (a + b - 1) / b; };
    auto gemm = [&](Lin l, const float* X, float* Y, int O, int C, int N, int act) {
        dim3 g(cdiv(N, 128), cdiv(O, 16));
        k_gemm<<<g, 128, 0, stream>>>(l.w, X, l.b, Y, O, C, N, act);
    };
    auto mlpL = [&](Lin l, const float* X, float* Y, int O, int C, int N) {
        gemm(l, X, Y, O, C, N, 0);
        k_gn_lrelu<<<8, 256, 0, stream>>>(Y, O, N);
    };

    // ================= SA stages =================
    const float* coordsIn[4]  = {pc, c1, c2, c3};
    float*       coordsOut[4] = {c1, c2, c3, c4};
    const int   SA_Np[4] = {4096, 1024, 256, 64};
    const int   SA_M[4]  = {1024, 256, 64, 16};
    const float SA_R2[4] = {0.01f, 0.04f, 0.16f, 0.64f};
    const int sa_conv_dims[4][4] = {{6, 32, 32, 0}, {64, 64, 64, 64},
                                    {128, 128, 128, 128}, {0, 0, 0, 0}};
    const int sa_mlp_dims[4][4]  = {{35, 32, 64, 0}, {67, 64, 128, 0},
                                    {131, 128, 256, 0}, {259, 128, 128, 128}};
    const int K = 32;

    const float* feat = pc;   // 6 x 4096
    for (int s = 0; s < 4; ++s) {
        int Np = SA_Np[s], M = SA_M[s];
        float* pp[2] = {T0, T1};
        const float* src = feat;
        for (int k = 0; k < sa_nconv[s]; ++k) {
            mlpL(sac[s][k], src, pp[k & 1], sa_conv_dims[s][k + 1], sa_conv_dims[s][k], Np);
            src = pp[k & 1];
        }
        k_centers<<<cdiv(M, 128), 128, 0, stream>>>(coordsIn[s], coordsOut[s], Np, M);
        k_knn<<<M, 256, 0, stream>>>(coordsIn[s], coordsOut[s], idxg[s], Np, M, K, SA_R2[s]);
        // grouping buffer: a T buffer different from conv output
        float* gbuf = (src == T0) ? T1 : ((src == T1) ? T2 : T1);
        int Cg = sa_mlp_dims[s][0];           // feat channels + 3
        int MK = M * K;
        k_group<<<cdiv(Cg * MK, 256), 256, 0, stream>>>(src, coordsIn[s], coordsOut[s],
                                                        idxg[s], gbuf, Cg - 3, Np, M, K);
        float* m0 = (gbuf == T0) ? T1 : T0;
        float* m1 = (gbuf == T2) ? T1 : T2;
        float* mp[2] = {m0, m1};
        const float* ms = gbuf;
        int lastC = Cg;
        for (int k = 0; k < sa_nmlp[s]; ++k) {
            mlpL(sam[s][k], ms, mp[k & 1], sa_mlp_dims[s][k + 1], sa_mlp_dims[s][k], MK);
            ms = mp[k & 1];
            lastC = sa_mlp_dims[s][k + 1];
        }
        k_maxpool<<<cdiv(lastC * M, 256), 256, 0, stream>>>(ms, pooled[s], lastC, M, K);
        feat = pooled[s];
    }

    // ================= Attention (128 x 16) =================
    gemm(att_qkv, pooled[3], T0, 384, 128, 16, 0);
    k_attention<<<1, 256, 0, stream>>>(T0, T1);
    gemm(att_out, T1, T2, 128, 128, 16, 0);
    k_add<<<cdiv(128 * 16, 256), 256, 0, stream>>>(pooled[3], T2, attF, 128 * 16);

    // ================= FP stages =================
    const float* denseC[4]  = {c3, c2, c1, pc};
    const float* sparseC[4] = {c4, c3, c2, c1};
    const int Nd_[4] = {64, 256, 1024, 4096};
    const int Ns_[4] = {16, 64, 256, 1024};
    const float* skipF[4] = {pooled[2], pooled[1], pooled[0], rgbp};
    const int skipC[4] = {256, 128, 64, 3};
    const int fp_mlp_dims[4][4]  = {{384, 128, 128, 0}, {256, 128, 128, 0},
                                    {192, 128, 128, 0}, {131, 128, 128, 64}};
    const int fp_conv_dims[4][4] = {{128, 128, 128, 128}, {128, 128, 128, 128},
                                    {128, 128, 128, 0}, {64, 64, 64, 0}};

    const float* cur = attF;
    int curC = 128;
    for (int j = 0; j < 4; ++j) {
        int Nd = Nd_[j], Ns = Ns_[j];
        k_knn3<<<cdiv(Nd, 128), 128, 0, stream>>>(denseC[j], sparseC[j], idx3, w3, Nd, Ns);
        k_interp<<<cdiv(curC * Nd, 256), 256, 0, stream>>>(cur, idx3, w3, T0, curC, Nd, Ns);
        k_rowcopy<<<cdiv(skipC[j] * Nd, 256), 256, 0, stream>>>(skipF[j],
                                                                T0 + (size_t)curC * Nd,
                                                                skipC[j] * Nd);
        const float* src = T0;
        float* pp[2] = {T1, T2};
        int pb = 0;
        for (int k = 0; k < fp_nmlp[j]; ++k) {
            mlpL(fpm[j][k], src, pp[pb], fp_mlp_dims[j][k + 1], fp_mlp_dims[j][k], Nd);
            src = pp[pb]; pb ^= 1;
            curC = fp_mlp_dims[j][k + 1];
        }
        for (int k = 0; k < fp_nconv[j]; ++k) {
            mlpL(fpc[j][k], src, pp[pb], fp_conv_dims[j][k + 1], fp_conv_dims[j][k], Nd);
            src = pp[pb]; pb ^= 1;
            curC = fp_conv_dims[j][k + 1];
        }
        cur = src;
    }

    // ================= classifier =================
    float* dst1 = (cur == T1) ? T2 : T1;
    mlpL(cls0, cur, dst1, 128, 64, NP);
    gemm(cls1, dst1, featZ, ZD, 128, NP, 0);

    // ================= heads =================
    gemm(sc0, featZ, T0, 32, 64, NP, 1);
    gemm(sc1, T0, sraw, 3, 32, NP, 0);
    gemm(ro0, featZ, T0, 32, 64, NP, 1);
    gemm(ro1, T0, rraw, 4, 32, NP, 0);
    gemm(op0, featZ, T0, 32, 64, NP, 1);
    gemm(op1, T0, oraw, 1, 32, NP, 0);
    k_post<<<cdiv(NP, 256), 256, 0, stream>>>(sraw, rraw, oraw, scaleB, rotB, opacB, NP);

    // ================= render =================
    k_project<<<cdiv(NP, 256), 256, 0, stream>>>(pc, scaleB, rotB, opacB, vm, intr, gp, zk, NP);
    k_sortz<<<1, 1024, 0, stream>>>(zk, order, NP);
    float* outRGB = (float*)d_out;
    float* outF = outRGB + 3 * NPIX;
    k_composite<<<NPIX / 256, 256, 0, stream>>>(gp, order, pc, featZ, outRGB, outF, NP);
}
